// MultiHeadAttention_40080634806695
// MI455X (gfx1250) — matrix operand order repaired and checked by reading, not compiled, not run on hardware
//
#include <hip/hip_runtime.h>
#include <hip/hip_bf16.h>
#include <stddef.h>

// ---------------- types ----------------
typedef __attribute__((ext_vector_type(16))) __bf16 v16bf;
typedef __attribute__((ext_vector_type(8)))  __bf16 v8bf;
typedef __attribute__((ext_vector_type(8)))  float  v8f;

#define B_    2
#define T_    2048
#define DIN_  1024
#define DOUT_ 1024
#define H_    16
#define HD_   64
#define BT_   (B_*T_)         // 4096
#define BH_   (B_*H_)         // 32

__device__ __forceinline__ unsigned short to_bf16(float f) {
  unsigned int u = __float_as_uint(f);
  u += 0x7FFFu + ((u >> 16) & 1u);      // round-to-nearest-even
  return (unsigned short)(u >> 16);
}

// A fragment (16x32 MxK, bf16): per-lane two contiguous 16B chunks.
// element e: K = k0 + 8*(lane>=16) + (e&7) + 16*(e>=8), M = m0 + (lane&15)
__device__ __forceinline__ v16bf ldfrag_a(const unsigned short* base, int row,
                                          int ld, int k0) {
  const int lane = threadIdx.x & 31;
  const unsigned short* p = base + (size_t)row * ld + k0 + ((lane >> 4) << 3);
  v8bf lo = *(const v8bf*)p;
  v8bf hi = *(const v8bf*)(p + 16);
  return __builtin_shufflevector(lo, hi, 0,1,2,3,4,5,6,7,8,9,10,11,12,13,14,15);
}

// B fragment (32x16 KxN, bf16) from N-major storage stored[n][k] (k contiguous):
// element e: K = k0 + 8*(lane>=16) + (e&7) + 16*(e>=8), N = n0 + (lane&15) -> two 16B chunks/lane
__device__ __forceinline__ v16bf ldfrag_b(const unsigned short* base, int n0,
                                          int ld, int k0) {
  const int lane = threadIdx.x & 31;
  const unsigned short* p = base + (size_t)(n0 + (lane & 15)) * ld + k0 + ((lane >> 4) << 3);
  v8bf lo = *(const v8bf*)p;
  v8bf hi = *(const v8bf*)(p + 16);
  return __builtin_shufflevector(lo, hi, 0,1,2,3,4,5,6,7,8,9,10,11,12,13,14,15);
}

__device__ __forceinline__ v8f wmma_bf16(v16bf a, v16bf b, v8f c) {
  return __builtin_amdgcn_wmma_f32_16x16x32_bf16(false, a, false, b, (short)0, c,
                                                 false, false);
}

// ---------------- fp32 -> bf16 convert (vectorized x4) ----------------
__global__ __launch_bounds__(256) void k_cvt4(const float* __restrict__ src,
                                              unsigned short* __restrict__ dst,
                                              int n) {
  int i = (blockIdx.x * 256 + threadIdx.x) * 4;
  if (i + 3 < n) {
    float4 v = *(const float4*)(src + i);
    dst[i + 0] = to_bf16(v.x);
    dst[i + 1] = to_bf16(v.y);
    dst[i + 2] = to_bf16(v.z);
    dst[i + 3] = to_bf16(v.w);
  }
}

// ---------------- QKV projection GEMM ----------------
// out[i,o] = sum_k x[i,k] * W[o,k]   (M=4096, N=1024, K=1024)
// Wave tile 64x128 (4 M-tiles x 8 N-tiles): 24 b128 loads -> 32 WMMAs per K-step.
// mode 0: store as [b,h,t,hd] bf16 (Q and K);  mode 1: store as [b,h,hd,t] bf16 (V^T)
__global__ __launch_bounds__(128) void k_proj(const unsigned short* __restrict__ xb,
                                              const unsigned short* __restrict__ wb,
                                              unsigned short* __restrict__ dst,
                                              int mode) {
  const int lane = threadIdx.x & 31;
  const int wave = threadIdx.x >> 5;
  const int nl = lane & 15, hl = lane >> 4;
  const int M0 = blockIdx.x * 128 + (wave >> 1) * 64;
  const int N0 = blockIdx.y * 256 + (wave & 1) * 128;

  const v8f zero = {0.f, 0.f, 0.f, 0.f, 0.f, 0.f, 0.f, 0.f};
  v8f c[4][8];
#pragma unroll
  for (int mt = 0; mt < 4; ++mt)
#pragma unroll
    for (int nt = 0; nt < 8; ++nt) c[mt][nt] = zero;

  for (int k0 = 0; k0 < DIN_; k0 += 32) {
    v16bf a[4], b[8];
#pragma unroll
    for (int mt = 0; mt < 4; ++mt)
      a[mt] = ldfrag_a(xb, M0 + mt * 16 + nl, DIN_, k0);
#pragma unroll
    for (int nt = 0; nt < 8; ++nt)
      b[nt] = ldfrag_b(wb, N0 + nt * 16, DIN_, k0);
#pragma unroll
    for (int mt = 0; mt < 4; ++mt)
#pragma unroll
      for (int nt = 0; nt < 8; ++nt)
        c[mt][nt] = wmma_bf16(a[mt], b[nt], c[mt][nt]);
  }

#pragma unroll
  for (int mt = 0; mt < 4; ++mt)
#pragma unroll
    for (int nt = 0; nt < 8; ++nt)
#pragma unroll
      for (int r = 0; r < 8; ++r) {
        int i = M0 + mt * 16 + r + 8 * hl;   // token index 0..4095
        int o = N0 + nt * 16 + nl;           // output neuron 0..1023
        int bb = i >> 11, t = i & (T_ - 1);
        int h = o >> 6, d = o & (HD_ - 1);
        size_t addr;
        if (mode == 0)
          addr = ((size_t)(bb * H_ + h) * T_ + t) * HD_ + d;   // [b,h,t,hd]
        else
          addr = ((size_t)(bb * H_ + h) * HD_ + d) * T_ + t;   // [b,h,hd,t]
        dst[addr] = to_bf16(c[mt][nt][r]);
      }
}

// ---------------- causal flash attention ----------------
// Q,K: [bh, T, 64] bf16 ; Vt: [bh, 64, T] bf16 ; out: [bh, 64, T] fp32 (= ref reshape)
// 64 queries per wave (4 M-tiles) so each K/V fragment load feeds 4 WMMAs.
#define P_LD 80   // padded LDS row stride (bf16 elems); 160B keeps b128 alignment
__global__ __launch_bounds__(128) void k_attn(const unsigned short* __restrict__ Qb,
                                              const unsigned short* __restrict__ Kb,
                                              const unsigned short* __restrict__ Vtb,
                                              float* __restrict__ out) {
  __shared__ unsigned short Pbuf[4][64 * P_LD];
  const int lane = threadIdx.x & 31;
  const int wave = threadIdx.x >> 5;
  const int nl = lane & 15, hl = lane >> 4;
  const int bh = blockIdx.y;
  const int i0 = (blockIdx.x * 4 + wave) * 64;

  const unsigned short* Qh = Qb  + (size_t)bh * T_ * HD_;
  const unsigned short* Kh = Kb  + (size_t)bh * T_ * HD_;
  const unsigned short* Vh = Vtb + (size_t)bh * HD_ * T_;
  float*                Oh = out + (size_t)bh * HD_ * T_;
  unsigned short* Pw = Pbuf[wave];

  // resident Q fragments: 4 M-tiles x 2 K-chunks (hd = 64 = 2*32)
  v16bf aq[4][2];
#pragma unroll
  for (int mt = 0; mt < 4; ++mt)
#pragma unroll
    for (int kc = 0; kc < 2; ++kc)
      aq[mt][kc] = ldfrag_a(Qh, i0 + mt * 16 + nl, HD_, kc * 32);

  const v8f zero = {0.f, 0.f, 0.f, 0.f, 0.f, 0.f, 0.f, 0.f};
  v8f ctx[4][4];
#pragma unroll
  for (int mt = 0; mt < 4; ++mt)
#pragma unroll
    for (int dt = 0; dt < 4; ++dt) ctx[mt][dt] = zero;

  float ms[32], ls[32];
#pragma unroll
  for (int q = 0; q < 32; ++q) { ms[q] = -3.0e38f; ls[q] = 0.0f; }

  const float SC = 0.18033688011112042f;  // (1/sqrt(64)) * log2(e)
  const int jend = i0 + 64;               // causal: key blocks with j0 <= i0+63

  for (int j0 = 0; j0 < jend; j0 += 64) {
    if (j0 + 64 < jend) {                 // prefetch next K/V key block
      __builtin_prefetch(Kh + (size_t)(j0 + 64 + lane) * HD_, 0, 1);
      __builtin_prefetch(Vh + (size_t)(lane * 2) * T_ + j0 + 64, 0, 1);
    }

    // ---- S = Q * K^T for this 64x64 block ----
    v8f s[4][4];
#pragma unroll
    for (int nt = 0; nt < 4; ++nt) {
      v16bf bk0 = ldfrag_b(Kh, j0 + nt * 16, HD_, 0);
      v16bf bk1 = ldfrag_b(Kh, j0 + nt * 16, HD_, 32);
#pragma unroll
      for (int mt = 0; mt < 4; ++mt) {
        v8f acc = wmma_bf16(aq[mt][0], bk0, zero);
        acc = wmma_bf16(aq[mt][1], bk1, acc);
        s[mt][nt] = acc;
      }
    }

    // ---- causal mask (only the diagonal block needs it) ----
    if (j0 + 64 > i0) {
#pragma unroll
      for (int mt = 0; mt < 4; ++mt)
#pragma unroll
        for (int nt = 0; nt < 4; ++nt)
#pragma unroll
          for (int r = 0; r < 8; ++r) {
            int qi = i0 + mt * 16 + r + 8 * hl;
            int jj = j0 + nt * 16 + nl;
            if (jj > qi) s[mt][nt][r] = -3.0e38f;
          }
    }

    // ---- online softmax; write P (bf16) to LDS ----
#pragma unroll
    for (int mt = 0; mt < 4; ++mt) {
#pragma unroll
      for (int r = 0; r < 8; ++r) {
        float vmax = fmaxf(fmaxf(s[mt][0][r], s[mt][1][r]),
                           fmaxf(s[mt][2][r], s[mt][3][r]));
        vmax = fmaxf(vmax, __shfl_xor(vmax, 1));
        vmax = fmaxf(vmax, __shfl_xor(vmax, 2));
        vmax = fmaxf(vmax, __shfl_xor(vmax, 4));
        vmax = fmaxf(vmax, __shfl_xor(vmax, 8));
        const int idx = mt * 8 + r;
        float mnew = fmaxf(ms[idx], vmax);
        float alpha = exp2f((ms[idx] - mnew) * SC);
        ms[idx] = mnew;
        const int row = mt * 16 + r + 8 * hl;
        float rs = 0.0f;
#pragma unroll
        for (int nt = 0; nt < 4; ++nt) {
          float p = exp2f((s[mt][nt][r] - mnew) * SC);
          rs += p;
          Pw[row * P_LD + nt * 16 + nl] = to_bf16(p);
        }
        rs += __shfl_xor(rs, 1);
        rs += __shfl_xor(rs, 2);
        rs += __shfl_xor(rs, 4);
        rs += __shfl_xor(rs, 8);
        ls[idx] = ls[idx] * alpha + rs;
#pragma unroll
        for (int dt = 0; dt < 4; ++dt) ctx[mt][dt][r] *= alpha;
      }
    }

    // cross-lane LDS dependency inside the wave: drain DS counter
    asm volatile("s_wait_dscnt 0" ::: "memory");

    // ---- ctx += P * V ----
    v16bf ap[4][2];
#pragma unroll
    for (int mt = 0; mt < 4; ++mt)
#pragma unroll
      for (int ks = 0; ks < 2; ++ks)
        ap[mt][ks] = ldfrag_a(Pw, mt * 16 + nl, P_LD, ks * 32);
#pragma unroll
    for (int dt = 0; dt < 4; ++dt)
#pragma unroll
      for (int ks = 0; ks < 2; ++ks) {
        v16bf bv = ldfrag_b(Vh, dt * 16, T_, j0 + ks * 32);
#pragma unroll
        for (int mt = 0; mt < 4; ++mt)
          ctx[mt][dt] = wmma_bf16(ap[mt][ks], bv, ctx[mt][dt]);
      }
  }

  // ---- epilogue: out[bh][d][t] = ctx / l  (matches reference reshape) ----
#pragma unroll
  for (int mt = 0; mt < 4; ++mt)
#pragma unroll
    for (int dt = 0; dt < 4; ++dt)
#pragma unroll
      for (int r = 0; r < 8; ++r) {
        int qi = i0 + mt * 16 + r + 8 * hl;
        int d = dt * 16 + nl;
        Oh[(size_t)d * T_ + qi] = ctx[mt][dt][r] / ls[mt * 8 + r];
      }
}

// ---------------- launch ----------------
extern "C" void kernel_launch(void* const* d_in, const int* in_sizes, int n_in,
                              void* d_out, int out_size, void* d_ws, size_t ws_size,
                              hipStream_t stream) {
  const float* x  = (const float*)d_in[0];
  const float* Wq = (const float*)d_in[1];
  const float* Wk = (const float*)d_in[2];
  const float* Wv = (const float*)d_in[3];
  float* out = (float*)d_out;

  unsigned short* w = (unsigned short*)d_ws;
  unsigned short* xb  = w;                    // 4096*1024
  unsigned short* wqb = w + 4194304;          // 1024*1024
  unsigned short* wkb = w + 5242880;
  unsigned short* wvb = w + 6291456;
  unsigned short* Qb  = w + 7340032;          // [bh, T, 64]
  unsigned short* Kb  = w + 11534336;         // [bh, T, 64]
  unsigned short* Vtb = w + 15728640;         // [bh, 64, T]

  // fp32 -> bf16
  k_cvt4<<<4096, 256, 0, stream>>>(x,  xb,  BT_ * DIN_);
  k_cvt4<<<1024, 256, 0, stream>>>(Wq, wqb, DOUT_ * DIN_);
  k_cvt4<<<1024, 256, 0, stream>>>(Wk, wkb, DOUT_ * DIN_);
  k_cvt4<<<1024, 256, 0, stream>>>(Wv, wvb, DOUT_ * DIN_);

  // projections (WG tile 128x256, 4 waves of 64x128)
  dim3 pgrid(BT_ / 128, DOUT_ / 256);
  k_proj<<<pgrid, 128, 0, stream>>>(xb, wqb, Qb, 0);
  k_proj<<<pgrid, 128, 0, stream>>>(xb, wkb, Kb, 0);
  k_proj<<<pgrid, 128, 0, stream>>>(xb, wvb, Vtb, 1);

  // causal attention: 256 queries per WG (64/wave), one (b,h) per blockIdx.y
  dim3 agrid(T_ / 256, BH_);
  k_attn<<<agrid, 128, 0, stream>>>(Qb, Kb, Vtb, out);
}
